// CrossAttention_83743272337486
// MI455X (gfx1250) — matrix operand order repaired and checked by reading, not compiled, not run on hardware
//
#include <hip/hip_runtime.h>

typedef __attribute__((ext_vector_type(16))) __bf16 bf16x16;
typedef __attribute__((ext_vector_type(8)))  __bf16 bf16x8;
typedef __attribute__((ext_vector_type(8)))  float  f32x8;
typedef __attribute__((ext_vector_type(4)))  unsigned int u32x4;
typedef __attribute__((ext_vector_type(8)))  int    i32x8;
typedef __attribute__((ext_vector_type(4)))  int    i32x4;

#define B_  8
#define L_  2048
#define D_  1024
#define NQROW (B_*L_)          // 16384 q rows; k rows are the next 16384
#define SCALE_ 0.03125f        // 1/sqrt(1024)
#define KPITCH 1040            // bf16 elems per staged kv row (TDM pad layout)
#define BUFH   (32*KPITCH)     // halves per kv buffer (66560 B)
#define SMEM_BYTES (2*BUFH*2 + 512*4 + 48*4)   // 2 kv bufs + Sred + stats = 135360

static __device__ __forceinline__
f32x8 wmma_bf16(bf16x16 a, bf16x16 b, f32x8 c) {
    return __builtin_amdgcn_wmma_f32_16x16x32_bf16(false, a, false, b,
                                                   (short)0, c, false, false);
}

// TDM pad layout: after every 256 DWORDs (512 halves) stored, 4 DWORDs
// (8 halves) of padding are inserted -> column c lives at c + (c>>9)*8.
static __device__ __forceinline__ int koff(int c) { return c + ((c >> 9) << 3); }

// Issue one TDM descriptor: load a 32(row) x 1024(col) bf16 tile, row stride
// 1024 elems, into LDS at lds_byte with hardware padding as above.
static __device__ __forceinline__
void tdm_load_tile(const __bf16* gsrc, unsigned int lds_byte)
{
    const unsigned long long ga = (unsigned long long)(uintptr_t)gsrc;
    u32x4 g0;
    g0[0] = 1u;                                   // count=1, user mode, no gather
    g0[1] = lds_byte;                             // LDS destination address
    g0[2] = (unsigned int)ga;                     // global_addr[31:0]
    g0[3] = (unsigned int)(ga >> 32) | (2u << 30); // global_addr[56:32] | type=2
    i32x8 g1;
    g1[0] = (int)((1u << 16)    // data_size = 2 bytes
                | (1u << 20)    // pad_enable
                | (7u << 22)    // pad_interval = 256 DWORDs
                | (3u << 25));  // pad_amount   = 4 DWORDs
    g1[1] = (int)(1024u << 16); // tensor_dim0 = 1024 (low 16 bits)
    g1[2] = (int)(0x8000u << 16); // tensor_dim0 hi=0 | tensor_dim1 = 32768 (lo)
    g1[3] = (int)(1024u << 16); // tensor_dim1 hi=0 | tile_dim0 = 1024
    g1[4] = 32;                 // tile_dim1 = 32 rows, tile_dim2 = 0
    g1[5] = 1024;               // tensor_dim0_stride[31:0]
    g1[6] = 0;                  // stride hi | tensor_dim1_stride lo
    g1[7] = 0;
    i32x4 gz4 = {};
    i32x8 gz8 = {};
    // amdgpu-toolchain (clang-23) 6-arg form:
    // (u32x4 g0, i32x8 g1, i32x4 g2, i32x4 g3, i32x8, i32 cpol)
    __builtin_amdgcn_tensor_load_to_lds(g0, g1, gz4, gz4, gz8, 0);
}

// ---------------- Kernel 1: Q/K projection, C = [x;enc] * Wq^T -> bf16 -----
__global__ __launch_bounds__(256)
void proj_kernel(const float* __restrict__ x, const float* __restrict__ enc,
                 const float* __restrict__ Wq, __bf16* __restrict__ proj)
{
    const int lane = threadIdx.x & 31;
    const int wave = threadIdx.x >> 5;
    const int job  = blockIdx.x * 8 + wave;   // 32768 jobs = 2048 mtiles * 16 n-groups
    const int mt   = job >> 4;
    const int n4   = job & 15;
    const int m    = lane & 15;
    const int half = lane >> 4;

    const int rowg = mt * 16 + m;
    const float* arow = (rowg < NQROW)
        ? (x   + (size_t)rowg * D_)
        : (enc + (size_t)(rowg - NQROW) * D_);

    f32x8 z = {};
    f32x8 acc[4] = {z, z, z, z};

    for (int d0 = 0; d0 < D_; d0 += 32) {
        if (d0 + 32 < D_) __builtin_prefetch(arow + d0 + 32, 0, 3);
        // A fragment (16x32): half=0 holds K 0-15, half=1 holds 16-31 (contiguous)
        bf16x16 A;
        f32x8 a0 = *(const f32x8*)(arow + d0 + half * 16);
        f32x8 a1 = *(const f32x8*)(arow + d0 + half * 16 + 8);
        #pragma unroll
        for (int j = 0; j < 8; ++j) { A[j] = (__bf16)a0[j]; A[8 + j] = (__bf16)a1[j]; }
        #pragma unroll
        for (int t = 0; t < 4; ++t) {
            // B fragment (32x16) of Wq^T: element e <-> K = half*16 + e (contiguous)
            const float* wrow = Wq + (size_t)(n4 * 64 + t * 16 + m) * D_ + d0 + half * 16;
            f32x8 b0 = *(const f32x8*)(wrow);
            f32x8 b1 = *(const f32x8*)(wrow + 8);
            bf16x16 Bf;
            #pragma unroll
            for (int j = 0; j < 8; ++j) { Bf[j] = (__bf16)b0[j]; Bf[8 + j] = (__bf16)b1[j]; }
            acc[t] = wmma_bf16(A, Bf, acc[t]);
        }
    }
    #pragma unroll
    for (int t = 0; t < 4; ++t) {
        #pragma unroll
        for (int r = 0; r < 8; ++r) {
            const int M = r + (half ? 8 : 0);
            proj[(size_t)(mt * 16 + M) * D_ + n4 * 64 + t * 16 + m] = (__bf16)acc[t][r];
        }
    }
}

// ---------------- Kernel 2: single-pass causal flash attention -------------
// Block (8 waves) owns one 16-row q-tile. Waves split D into 128-dim slices:
// partial scores -> LDS ds_add reduction -> block online softmax -> P(bf16)@V
// per-wave H slice. K/V tiles double-buffered in LDS, filled by the TDM.
__global__ __launch_bounds__(256)
void flash_kernel(const __bf16* __restrict__ proj, float* __restrict__ out)
{
    extern __shared__ __attribute__((aligned(128))) char smem[];
    __bf16* kv   = (__bf16*)smem;                       // 2 x (32 x KPITCH) bf16
    float*  Sred = (float*)(smem + 2 * BUFH * 2);       // 2 x 16 x 16 f32
    float*  fL   = Sred + 512;
    float*  mL   = fL + 16;
    float*  iL   = mL + 16;

    const int b  = blockIdx.x >> 7;
    const int qt = blockIdx.x & 127;
    const int tid = threadIdx.x;
    const int wave = tid >> 5, lane = tid & 31;
    const int m = lane & 15, half = lane >> 4;
    const int dws = wave * 128;                         // this wave's D/H slice

    const int qrow0 = b * L_ + qt * 16;
    const __bf16* qp = proj + (size_t)(qrow0 + m) * D_;
    const __bf16* kp = proj + (size_t)(NQROW + b * L_) * D_;
    const unsigned int lds0 = (unsigned int)(uintptr_t)kv;

    // Preload this wave's 4 q A-fragments (16 x 32 along its D slice)
    bf16x16 QF[4];
    #pragma unroll
    for (int ds = 0; ds < 4; ++ds) {
        const int d0 = dws + ds * 32;
        bf16x8 a0 = *(const bf16x8*)(qp + d0 + half * 16);
        bf16x8 a1 = *(const bf16x8*)(qp + d0 + half * 16 + 8);
        #pragma unroll
        for (int j = 0; j < 8; ++j) { QF[ds][j] = a0[j]; QF[ds][8 + j] = a1[j]; }
    }

    f32x8 z = {};
    f32x8 O[8] = {z, z, z, z, z, z, z, z};
    float mrun = -1e30f, srun = 0.0f;                   // live in lanes tid<16

    const int nstep = (qt * 16 + 16 + 31) >> 5;         // 32-key steps, causal

    if (tid < 32) tdm_load_tile(kp, lds0);              // prologue: TDM step 0

    for (int s = 0; s < nstep; ++s) {
        const int k0 = s * 32;
        const __bf16* kvb = kv + (s & 1) * BUFH;

        if (tid < 32) __builtin_amdgcn_s_wait_tensorcnt(0);  // buf[s&1] filled
        __syncthreads();   // (A) tile visible; prev readers of buf[(s+1)&1] done

        if (tid < 32 && s + 1 < nstep)                  // overlap next DMA
            tdm_load_tile(kp + (size_t)(k0 + 32) * D_,
                          lds0 + (unsigned int)(((s + 1) & 1) * (BUFH * 2)));
        Sred[tid] = 0.0f; Sred[tid + 256] = 0.0f;
        __syncthreads();   // (B)

        // Partial scores over dims [dws, dws+128): two 16x16 key sub-tiles
        f32x8 S0 = z, S1 = z;
        #pragma unroll
        for (int ds = 0; ds < 4; ++ds) {
            const int cb = koff(dws + ds * 32 + half * 16);
            const __bf16* r0 = kvb + (size_t)m * KPITCH + cb;
            const __bf16* r1 = kvb + (size_t)(m + 16) * KPITCH + cb;
            bf16x8 b0a = *(const bf16x8*)(r0), b0b = *(const bf16x8*)(r0 + 8);
            bf16x8 b1a = *(const bf16x8*)(r1), b1b = *(const bf16x8*)(r1 + 8);
            bf16x16 B0, B1;
            #pragma unroll
            for (int j = 0; j < 8; ++j) {
                B0[j] = b0a[j]; B0[8 + j] = b0b[j];
                B1[j] = b1a[j]; B1[8 + j] = b1b[j];
            }
            S0 = wmma_bf16(QF[ds], B0, S0);
            S1 = wmma_bf16(QF[ds], B1, S1);
        }
        // Cross-wave split-D reduction (ds_add_f32)
        #pragma unroll
        for (int r = 0; r < 8; ++r) {
            const int M = r + (half ? 8 : 0);
            atomicAdd(&Sred[M * 16 + m], S0[r]);
            atomicAdd(&Sred[256 + M * 16 + m], S1[r]);
        }
        __syncthreads();   // (C)

        // Online softmax stats, one thread per q-row
        if (tid < 16) {
            const int qglob = qt * 16 + tid;
            float tmax = -1e30f;
            #pragma unroll
            for (int c = 0; c < 32; ++c)
                if (k0 + c <= qglob) {
                    float v = Sred[(c >> 4) * 256 + tid * 16 + (c & 15)] * SCALE_;
                    tmax = fmaxf(tmax, v);
                }
            const float mnew = fmaxf(mrun, tmax);
            const float fac  = __expf(mrun - mnew);
            float ssum = 0.0f;
            #pragma unroll
            for (int c = 0; c < 32; ++c)
                if (k0 + c <= qglob) {
                    float v = Sred[(c >> 4) * 256 + tid * 16 + (c & 15)] * SCALE_;
                    ssum += __expf(v - mnew);
                }
            srun = srun * fac + ssum;
            mrun = mnew;
            fL[tid] = fac; mL[tid] = mnew;
        }
        __syncthreads();   // (D)

        // Build P (A fragment, 16 x 32 keys) in bf16
        bf16x16 P;
        {
            const float mn = mL[m];
            const int qglob = qt * 16 + m;
            #pragma unroll
            for (int e = 0; e < 16; ++e) {
                const int kk = half * 16 + e;
                float p = 0.0f;
                if (k0 + kk <= qglob) {
                    float v = Sred[(kk >> 4) * 256 + m * 16 + (kk & 15)] * SCALE_;
                    p = __expf(v - mn);
                }
                P[e] = (__bf16)p;
            }
        }
        // Rescale O, then O += P @ V over this wave's 128-col H slice
        float fr[8];
        #pragma unroll
        for (int r = 0; r < 8; ++r) fr[r] = fL[r + (half ? 8 : 0)];
        #pragma unroll
        for (int t = 0; t < 8; ++t)
            #pragma unroll
            for (int r = 0; r < 8; ++r) O[t][r] *= fr[r];

        #pragma unroll
        for (int t = 0; t < 8; ++t) {
            bf16x16 Bv;   // V B-fragment: K = key row (half*16+e), N col = h
            const int hc = koff(dws + t * 16 + m);
            #pragma unroll
            for (int e = 0; e < 16; ++e)
                Bv[e] = kvb[(size_t)(half * 16 + e) * KPITCH + hc];
            O[t] = wmma_bf16(P, Bv, O[t]);
        }
    }

    __syncthreads();
    if (tid < 16) iL[tid] = 1.0f / srun;
    __syncthreads();
    float iv[8];
    #pragma unroll
    for (int r = 0; r < 8; ++r) iv[r] = iL[r + (half ? 8 : 0)];
    #pragma unroll
    for (int t = 0; t < 8; ++t) {
        const int h = dws + t * 16 + m;
        #pragma unroll
        for (int r = 0; r < 8; ++r)
            out[(size_t)(qrow0 + r + (half ? 8 : 0)) * D_ + h] = O[t][r] * iv[r];
    }
}

extern "C" void kernel_launch(void* const* d_in, const int* in_sizes, int n_in,
                              void* d_out, int out_size, void* d_ws, size_t ws_size,
                              hipStream_t stream) {
    (void)in_sizes; (void)n_in; (void)out_size; (void)ws_size;
    const float* enc = (const float*)d_in[0];   // enc_outputs [8,2048,1024] f32
    const float* x   = (const float*)d_in[1];   // x           [8,2048,1024] f32
    const float* Wq  = (const float*)d_in[2];   // Wq          [1024,1024]   f32
    float* out = (float*)d_out;

    // Workspace: bf16 projections, q rows then k rows: 32768 x 1024 bf16 = 64 MB
    __bf16* proj = (__bf16*)d_ws;

    proj_kernel<<<4096, 256, 0, stream>>>(x, enc, Wq, proj);
    flash_kernel<<<1024, 256, SMEM_BYTES, stream>>>(proj, out);
}
